// EvaBlock_20890720928478
// MI455X (gfx1250) — matrix operand order repaired and checked by reading, not compiled, not run on hardware
//
#include <hip/hip_runtime.h>

#define NH 12
#define HD 64
#define EVA_EPS 1e-5f
#define LNRM_EPS 1e-5f

typedef __attribute__((ext_vector_type(16))) __bf16 v16bf;
typedef __attribute__((ext_vector_type(8)))  float  v8f;

static __device__ __forceinline__ v8f wmma_bf16(v16bf a, v16bf b, v8f c) {
  return __builtin_amdgcn_wmma_f32_16x16x32_bf16(false, a, false, b, (short)0, c,
                                                 false, false);
}

// ---- CDNA5 async global->LDS copy (ASYNCcnt tracked), 16B per lane ----
static __device__ __forceinline__ void g2l_async_b128(void* lds_ptr,
                                                      const void* gptr) {
  // low 32 bits of a flat shared pointer are the LDS byte offset
  unsigned int l = (unsigned int)(unsigned long long)lds_ptr;
  asm volatile("global_load_async_to_lds_b128 %0, %1, off"
               : : "v"(l), "v"(gptr) : "memory");
}

template <int N>
static __device__ __forceinline__ void wait_asynccnt() {
#if __has_builtin(__builtin_amdgcn_s_wait_asynccnt)
  __builtin_amdgcn_s_wait_asynccnt((short)N);
#else
  asm volatile("s_wait_asynccnt %0" : : "i"(N) : "memory");
#endif
}

// ---- WMMA fragment gathers from LDS (layouts per CDNA5 ISA 7.12.2) ----
// A 16x32, LDS fp32 row-major [m][k]; converts to bf16 during gather.
static __device__ __forceinline__ v16bf frag_a_mk_f32(const float* lds, int lda,
                                                      int m0, int k0, int lane) {
  int m  = m0 + (lane & 15);
  int kb = k0 + ((lane >> 4) << 3);      // lanes 16..31: K base +8
  v16bf a;
#pragma unroll
  for (int e = 0; e < 16; ++e)
    a[e] = (__bf16)lds[(size_t)m * lda + kb + ((e >> 3) << 4) + (e & 7)];
  return a;
}
// B 32x16 where LDS fp32 holds W as [n][k] (row = output col, contiguous k).
static __device__ __forceinline__ v16bf frag_b_nk_f32(const float* lds, int ldb,
                                                      int n0, int k0, int lane) {
  int n  = n0 + (lane & 15);
  int kb = k0 + ((lane >> 4) << 3);      // lanes 16..31: K base +8
  v16bf b;
#pragma unroll
  for (int e = 0; e < 16; ++e) b[e] = (__bf16)lds[(size_t)n * ldb + kb + ((e >> 3) << 4) + (e & 7)];
  return b;
}
// bf16-LDS variants (used by kv/attn kernels)
static __device__ __forceinline__ v16bf frag_a_mk(const __bf16* lds, int lda,
                                                  int m0, int k0, int lane) {
  int m  = m0 + (lane & 15);
  int kb = k0 + ((lane >> 4) << 3);
  v16bf a;
#pragma unroll
  for (int e = 0; e < 16; ++e)
    a[e] = lds[(size_t)m * lda + kb + ((e >> 3) << 4) + (e & 7)];
  return a;
}
static __device__ __forceinline__ v16bf frag_a_km(const __bf16* lds, int ldm,
                                                  int k0, int m0, int lane) {
  int m  = m0 + (lane & 15);
  int kb = k0 + ((lane >> 4) << 3);
  v16bf a;
#pragma unroll
  for (int e = 0; e < 16; ++e)
    a[e] = lds[(size_t)(kb + ((e >> 3) << 4) + (e & 7)) * ldm + m];
  return a;
}
static __device__ __forceinline__ v16bf frag_b_kn(const __bf16* lds, int ldn,
                                                  int k0, int n0, int lane) {
  int n  = n0 + (lane & 15);
  int kb = k0 + ((lane >> 4) << 3);
  v16bf b;
#pragma unroll
  for (int e = 0; e < 16; ++e) b[e] = lds[(size_t)(kb + ((e >> 3) << 4) + (e & 7)) * ldn + n];
  return b;
}

// =====================================================================
// Kernel: generic  out[M,Nc] = A[M,K] @ W[Nc,K]^T + bias   (fp32 io, bf16 math)
// block 256 thr = 8 waves; tile 128x64; wave tile 32x32; BK=32.
// Double-buffered LDS fp32 tiles filled by global_load_async_to_lds_b128,
// software-pipelined on ASYNCcnt.
// =====================================================================
#define GBM 128
#define GBN 64
#define GBK 32
#define GLDF 36   // padded fp32 LDS row stride (144B, 16B-aligned)

__global__ __launch_bounds__(256) void eva_gemm_bias_kernel(
    const float* __restrict__ A, const float* __restrict__ W,
    const float* __restrict__ bias, float* __restrict__ out,
    int M, int K, int Nc) {
  __shared__ float Af[2][GBM * GLDF];
  __shared__ float Wf[2][GBN * GLDF];
  const int tid  = threadIdx.x;
  const int lane = tid & 31;
  const int wave = tid >> 5;
  const int wm = wave >> 1, wn = wave & 1;
  const int mBase = blockIdx.x * GBM;
  const int nBase = blockIdx.y * GBN;

  const int col4 = (tid & 7) * 4;   // 8 float4 groups across BK=32
  const int row0 = tid >> 3;        // 0..31

  // 6 async b128 copies per thread per tile (4 for A, 2 for W)
  auto issue_tile = [&](int buf, int k0) {
#pragma unroll
    for (int rr = 0; rr < 4; ++rr) {
      int row = row0 + rr * 32;
      int gm  = mBase + row;
      if (gm >= M) gm = M - 1;   // clamp: garbage rows are never stored
      g2l_async_b128(&Af[buf][row * GLDF + col4],
                     A + (size_t)gm * K + k0 + col4);
    }
#pragma unroll
    for (int rr = 0; rr < 2; ++rr) {
      int row = row0 + rr * 32;
      g2l_async_b128(&Wf[buf][row * GLDF + col4],
                     W + (size_t)(nBase + row) * K + k0 + col4);
    }
  };

  v8f acc[2][2];
#pragma unroll
  for (int i = 0; i < 2; ++i)
#pragma unroll
    for (int j = 0; j < 2; ++j) acc[i][j] = (v8f){0,0,0,0,0,0,0,0};

  const int ntiles = K / GBK;
  issue_tile(0, 0);
  if (ntiles > 1) issue_tile(1, GBK);

  for (int t = 0; t < ntiles; ++t) {
    const int buf = t & 1;
    // tile t complete when only the (<=6) copies of tile t+1 remain in flight
    if (t + 1 < ntiles) wait_asynccnt<6>();
    else                wait_asynccnt<0>();
    __syncthreads();

    v16bf bf[2];
#pragma unroll
    for (int tn = 0; tn < 2; ++tn)
      bf[tn] = frag_b_nk_f32(Wf[buf], GLDF, wn * 32 + tn * 16, 0, lane);
#pragma unroll
    for (int tm = 0; tm < 2; ++tm) {
      v16bf af = frag_a_mk_f32(Af[buf], GLDF, wm * 32 + tm * 16, 0, lane);
#pragma unroll
      for (int tn = 0; tn < 2; ++tn)
        acc[tm][tn] = wmma_bf16(af, bf[tn], acc[tm][tn]);
    }
    __syncthreads();                 // everyone done reading buf
    if (t + 2 < ntiles) issue_tile(buf, (t + 2) * GBK);
  }

#pragma unroll
  for (int tm = 0; tm < 2; ++tm)
#pragma unroll
    for (int tn = 0; tn < 2; ++tn) {
      int n = nBase + wn * 32 + tn * 16 + (lane & 15);
      float bv = bias[n];
#pragma unroll
      for (int r = 0; r < 8; ++r) {
        int m = mBase + wm * 32 + tm * 16 + r + ((lane >> 4) << 3);
        if (m < M) out[(size_t)m * Nc + n] = acc[tm][tn][r] + bv;
      }
    }
}

// =====================================================================
// Kernel: fused combined qkv bias  [q_bias | 0 | v_bias]
// =====================================================================
__global__ void eva_build_bias_kernel(const float* __restrict__ qb,
                                      const float* __restrict__ vb,
                                      float* __restrict__ ob) {
  int i = blockIdx.x * blockDim.x + threadIdx.x;
  if (i >= 2304) return;
  float v = 0.f;
  if (i < 768) v = qb[i];
  else if (i >= 1536) v = vb[i - 1536];
  ob[i] = v;
}

// =====================================================================
// Kernel: interleaved RoPE + softmax feature map over D=64, in place on q,k.
// one wave32 per (token, head); lane holds the rotation pair (2l, 2l+1).
// =====================================================================
__global__ __launch_bounds__(256) void eva_rope_softmax_kernel(
    float* __restrict__ qkv, const float* __restrict__ rope, int B, int N) {
  int wid  = (blockIdx.x * blockDim.x + threadIdx.x) >> 5;
  int lane = threadIdx.x & 31;
  int total = B * N * NH;
  if (wid >= total) return;
  int h = wid % NH;
  int m = wid / NH;
  int n = m % N;
#pragma unroll
  for (int s = 0; s < 2; ++s) {   // s=0: q, s=1: k
    float* row = qkv + (size_t)m * 2304 + s * 768 + h * HD;
    float x0 = row[2 * lane], x1 = row[2 * lane + 1];
    if (n >= 1) {   // RoPE on non-prefix tokens
      const float* rp = rope + (size_t)(n - 1) * 128;
      float s0 = rp[2 * lane], s1 = rp[2 * lane + 1];
      float c0 = rp[64 + 2 * lane], c1 = rp[64 + 2 * lane + 1];
      float r0 = x0 * c0 - x1 * s0;   // x*cos + rot(x)*sin, rot=(-x1, x0)
      float r1 = x1 * c1 + x0 * s1;
      x0 = r0; x1 = r1;
    }
    float mx = fmaxf(x0, x1);
#pragma unroll
    for (int off = 16; off > 0; off >>= 1) mx = fmaxf(mx, __shfl_xor(mx, off, 32));
    float e0 = __expf(x0 - mx), e1 = __expf(x1 - mx);
    float sm = e0 + e1;
#pragma unroll
    for (int off = 16; off > 0; off >>= 1) sm += __shfl_xor(sm, off, 32);
    float inv = 1.f / sm;
    row[2 * lane]     = e0 * inv;
    row[2 * lane + 1] = e1 * inv;
  }
}

// =====================================================================
// Kernel: kv[b,h,d,e] = sum_n fk[n,d]*v[n,e]; ksum[b,h,d] = sum_n fk[n,d]
// block 128 thr = 4 waves; each wave owns a 16-row d-block x all 64 e cols.
// N split into chunks of 512 across blockIdx.y; f32 atomics to combine.
// =====================================================================
#define KVCHUNK 512
#define KVLD 72

__global__ __launch_bounds__(128) void eva_kv_ksum_kernel(
    const float* __restrict__ qkv, float* __restrict__ kv,
    float* __restrict__ ksum, int B, int N) {
  __shared__ __bf16 FK[32 * KVLD];
  __shared__ __bf16 VS[32 * KVLD];
  const int bh = blockIdx.x;
  const int b  = bh / NH, h = bh % NH;
  const int chunk0 = blockIdx.y * KVCHUNK;
  const int tid = threadIdx.x, lane = tid & 31, wave = tid >> 5;

  v8f acc[4];
#pragma unroll
  for (int t = 0; t < 4; ++t) acc[t] = (v8f){0,0,0,0,0,0,0,0};
  float ks = 0.f;

  const float* fkBase = qkv + (size_t)b * N * 2304 + 768 + h * HD;
  const float* vBase  = qkv + (size_t)b * N * 2304 + 1536 + h * HD;
  const int col4 = (tid & 15) * 4;
  const int row0 = tid >> 4;

  for (int kk = 0; kk < KVCHUNK; kk += 32) {
    __syncthreads();
#pragma unroll
    for (int rr = 0; rr < 4; ++rr) {
      int row = row0 + rr * 8;
      int n = chunk0 + kk + row;
      float4 fv = make_float4(0.f, 0.f, 0.f, 0.f);
      float4 vv = make_float4(0.f, 0.f, 0.f, 0.f);
      if (n < N) {
        fv = *(const float4*)(fkBase + (size_t)n * 2304 + col4);
        vv = *(const float4*)(vBase + (size_t)n * 2304 + col4);
      }
      __bf16* d1 = FK + row * KVLD + col4;
      d1[0]=(__bf16)fv.x; d1[1]=(__bf16)fv.y; d1[2]=(__bf16)fv.z; d1[3]=(__bf16)fv.w;
      __bf16* d2 = VS + row * KVLD + col4;
      d2[0]=(__bf16)vv.x; d2[1]=(__bf16)vv.y; d2[2]=(__bf16)vv.z; d2[3]=(__bf16)vv.w;
    }
    __syncthreads();
    if (tid < 64) {
#pragma unroll
      for (int r = 0; r < 32; ++r) ks += (float)FK[r * KVLD + tid];
    }
    v16bf af = frag_a_km(FK, KVLD, 0, wave * 16, lane);  // A = fk^T (d x n)
#pragma unroll
    for (int t = 0; t < 4; ++t) {
      v16bf bf = frag_b_kn(VS, KVLD, 0, t * 16, lane);   // B = v (n x e)
      acc[t] = wmma_bf16(af, bf, acc[t]);
    }
  }

  float* kvp = kv + (size_t)bh * HD * HD;
#pragma unroll
  for (int t = 0; t < 4; ++t) {
    int e = t * 16 + (lane & 15);
#pragma unroll
    for (int r = 0; r < 8; ++r) {
      int d = wave * 16 + r + ((lane >> 4) << 3);
      atomicAdd(kvp + (size_t)d * HD + e, acc[t][r]);
    }
  }
  if (tid < 64) atomicAdd(ksum + (size_t)bh * HD + tid, ks);
}

// =====================================================================
// Kernel: attn[m, h*64+e] = (fq @ kv)[n,e] / (fq·ksum + eps)
// block 256 thr = 8 waves; tile 128 rows x 64 cols (full head); K=64.
// grid: (ceil(N/128), B, H)
// =====================================================================
#define AOLD 72

__global__ __launch_bounds__(256) void eva_attn_out_kernel(
    const float* __restrict__ qkv, const float* __restrict__ kv,
    const float* __restrict__ ksum, float* __restrict__ attn, int B, int N) {
  __shared__ __bf16 FQ[128 * AOLD];
  __shared__ __bf16 KV[64 * AOLD];
  __shared__ float  den[128];
  const int b = blockIdx.y, h = blockIdx.z;
  const int bh = b * NH + h;
  const int nBase = blockIdx.x * 128;
  const int tid = threadIdx.x, lane = tid & 31, wave = tid >> 5;
  const int wm = wave >> 1, wn = wave & 1;

  const int col4 = (tid & 15) * 4;
  const int row0 = tid >> 4;
  const float* fqBase = qkv + (size_t)b * N * 2304 + h * HD;  // fq overwrote q
#pragma unroll
  for (int rr = 0; rr < 8; ++rr) {   // FQ tile: 128 x 64
    int row = row0 + rr * 16;
    int n = nBase + row;
    float4 v = make_float4(0.f, 0.f, 0.f, 0.f);
    if (n < N) v = *(const float4*)(fqBase + (size_t)n * 2304 + col4);
    __bf16* d = FQ + row * AOLD + col4;
    d[0]=(__bf16)v.x; d[1]=(__bf16)v.y; d[2]=(__bf16)v.z; d[3]=(__bf16)v.w;
  }
  const float* kvp = kv + (size_t)bh * HD * HD;
#pragma unroll
  for (int rr = 0; rr < 4; ++rr) {   // KV tile: 64 x 64
    int row = row0 + rr * 16;
    float4 v = *(const float4*)(kvp + (size_t)row * HD + col4);
    __bf16* d = KV + row * AOLD + col4;
    d[0]=(__bf16)v.x; d[1]=(__bf16)v.y; d[2]=(__bf16)v.z; d[3]=(__bf16)v.w;
  }
  __syncthreads();

  if (tid < 128) {   // per-row denom = fq · ksum + eps
    const float* kp = ksum + (size_t)bh * HD;
    float s = 0.f;
#pragma unroll
    for (int d = 0; d < HD; ++d) s += (float)FQ[tid * AOLD + d] * kp[d];
    den[tid] = s + EVA_EPS;
  }

  v8f acc[2][2];
#pragma unroll
  for (int i = 0; i < 2; ++i)
#pragma unroll
    for (int j = 0; j < 2; ++j) acc[i][j] = (v8f){0,0,0,0,0,0,0,0};

#pragma unroll
  for (int k0 = 0; k0 < HD; k0 += 32) {
    v16bf bf[2];
#pragma unroll
    for (int tn = 0; tn < 2; ++tn)
      bf[tn] = frag_b_kn(KV, AOLD, k0, wn * 32 + tn * 16, lane);
#pragma unroll
    for (int tm = 0; tm < 2; ++tm) {
      v16bf af = frag_a_mk(FQ, AOLD, wm * 32 + tm * 16, k0, lane);
#pragma unroll
      for (int tn = 0; tn < 2; ++tn) acc[tm][tn] = wmma_bf16(af, bf[tn], acc[tm][tn]);
    }
  }
  __syncthreads();

#pragma unroll
  for (int tm = 0; tm < 2; ++tm)
#pragma unroll
    for (int tn = 0; tn < 2; ++tn) {
      int e = wn * 32 + tn * 16 + (lane & 15);
#pragma unroll
      for (int r = 0; r < 8; ++r) {
        int lrow = wm * 32 + tm * 16 + r + ((lane >> 4) << 3);
        int n = nBase + lrow;
        if (n < N) {
          size_t m = (size_t)b * N + n;
          attn[m * 768 + h * HD + e] = acc[tm][tn][r] / den[lrow];
        }
      }
    }
}

// =====================================================================
// Kernel: in-place LayerNorm over C=768, one block per row
// =====================================================================
__global__ __launch_bounds__(256) void eva_layernorm_kernel(
    float* __restrict__ attn, const float* __restrict__ w,
    const float* __restrict__ bias, int M) {
  __shared__ float red[256];
  const int m = blockIdx.x;
  float* row = attn + (size_t)m * 768;
  float s = 0.f;
  for (int i = threadIdx.x; i < 768; i += 256) s += row[i];
  red[threadIdx.x] = s;
  __syncthreads();
  for (int off = 128; off > 0; off >>= 1) {
    if (threadIdx.x < off) red[threadIdx.x] += red[threadIdx.x + off];
    __syncthreads();
  }
  const float mu = red[0] * (1.f / 768.f);
  __syncthreads();
  float v = 0.f;
  for (int i = threadIdx.x; i < 768; i += 256) {
    float d = row[i] - mu;
    v += d * d;
  }
  red[threadIdx.x] = v;
  __syncthreads();
  for (int off = 128; off > 0; off >>= 1) {
    if (threadIdx.x < off) red[threadIdx.x] += red[threadIdx.x + off];
    __syncthreads();
  }
  const float rstd = rsqrtf(red[0] * (1.f / 768.f) + LNRM_EPS);
  for (int i = threadIdx.x; i < 768; i += 256)
    row[i] = (row[i] - mu) * rstd * w[i] + bias[i];
}

// =====================================================================
extern "C" void kernel_launch(void* const* d_in, const int* in_sizes, int n_in,
                              void* d_out, int out_size, void* d_ws, size_t ws_size,
                              hipStream_t stream) {
  const float* x      = (const float*)d_in[0];
  const float* rope   = (const float*)d_in[1];
  const float* qkv_w  = (const float*)d_in[2];
  const float* q_bias = (const float*)d_in[3];
  const float* v_bias = (const float*)d_in[4];
  const float* norm_w = (const float*)d_in[5];
  const float* norm_b = (const float*)d_in[6];
  const float* proj_w = (const float*)d_in[7];
  const float* proj_b = (const float*)d_in[8];
  float* out = (float*)d_out;

  const int N = in_sizes[1] / 128 + 1;          // rope = (N-1) x 128
  const int B = in_sizes[0] / (N * 768);
  const int M = B * N;

  float* ws    = (float*)d_ws;
  float* qkv   = ws;                                    // M * 2304
  float* kvb   = qkv + (size_t)M * 2304;                // B*H*64*64
  float* ksm   = kvb + (size_t)B * NH * HD * HD;        // B*H*64
  float* attn  = ksm + (size_t)B * NH * HD;             // M * 768
  float* qbias = attn + (size_t)M * 768;                // 2304

  // zero the atomic accumulators (kv + ksum are contiguous)
  hipMemsetAsync(kvb, 0,
                 ((size_t)B * NH * HD * HD + (size_t)B * NH * HD) * sizeof(float),
                 stream);

  eva_build_bias_kernel<<<(2304 + 255) / 256, 256, 0, stream>>>(q_bias, v_bias, qbias);

  // 1) qkv = x @ qkv_w^T + [qb|0|vb]
  eva_gemm_bias_kernel<<<dim3((M + GBM - 1) / GBM, 2304 / GBN), 256, 0, stream>>>(
      x, qkv_w, qbias, qkv, M, 768, 2304);

  // 2) RoPE + softmax feature map on q,k (in place)
  {
    int waves = M * NH;
    eva_rope_softmax_kernel<<<(waves + 7) / 8, 256, 0, stream>>>(qkv, rope, B, N);
  }

  // 3) kv / ksum reductions (N split + atomics)
  eva_kv_ksum_kernel<<<dim3(B * NH, (N + KVCHUNK - 1) / KVCHUNK), 128, 0, stream>>>(
      qkv, kvb, ksm, B, N);

  // 4) out = fq @ kv / (fq·ksum + eps), write [B,N,C]
  eva_attn_out_kernel<<<dim3((N + 127) / 128, B, NH), 256, 0, stream>>>(
      qkv, kvb, ksm, attn, B, N);

  // 5) LayerNorm (in place) then proj GEMM
  eva_layernorm_kernel<<<M, 256, 0, stream>>>(attn, norm_w, norm_b, M);
  eva_gemm_bias_kernel<<<dim3((M + GBM - 1) / GBM, 768 / GBN), 256, 0, stream>>>(
      attn, proj_w, proj_b, out, M, 768, 768);
}
